// HomoGATv2Encoder_15805479649924
// MI455X (gfx1250) — matrix operand order repaired and checked by reading, not compiled, not run on hardware
//
#include <hip/hip_runtime.h>
#include <hip/hip_bf16.h>
#include <math.h>

// ---------------------------------------------------------------------------
// Problem constants (match reference)
// ---------------------------------------------------------------------------
#define NN    100000   // nodes
#define EE    800000   // edges
#define FDIM  256      // feature dim (F_IN == HID == 256)
#define HEADS 8
#define CDIM  32       // per-head dim
#define OUTD  128
#define LEAKY 0.2f

typedef __bf16 bf16_t;
typedef __attribute__((ext_vector_type(16))) __bf16 v16bf;
typedef __attribute__((ext_vector_type(8)))  __bf16 v8bf;
typedef __attribute__((ext_vector_type(8)))  float  v8f;
typedef __attribute__((ext_vector_type(4)))  unsigned int uint32x4;
typedef __attribute__((ext_vector_type(8)))  int int32x8;
typedef __attribute__((ext_vector_type(4)))  int int32x4;

#if defined(__has_builtin)
#  if __has_builtin(__builtin_amdgcn_tensor_load_to_lds)
#    define HAVE_TDM 1
#  endif
#endif

static __device__ __forceinline__ bf16_t f2bf(float f) {
    __hip_bfloat16 h = __float2bfloat16(f);
    bf16_t r;
    __builtin_memcpy(&r, &h, 2);
    return r;
}

static __device__ __forceinline__ float lrelu(float v) {
    return v > 0.f ? v : LEAKY * v;
}

// float atomic-max via ordered-int trick (valid with -inf init)
static __device__ __forceinline__ void atomicMaxF(float* addr, float val) {
    if (val >= 0.f) atomicMax((int*)addr, __float_as_int(val));
    else            atomicMin((unsigned int*)addr, (unsigned int)__float_as_int(val));
}

// ---------------------------------------------------------------------------
// Utility kernels
// ---------------------------------------------------------------------------
__global__ void k_fill(float* __restrict__ p, float v, int n) {
    int i = blockIdx.x * blockDim.x + threadIdx.x;
    if (i < n) p[i] = v;
}

__global__ void k_f32_to_bf16(const float* __restrict__ in, bf16_t* __restrict__ out, int n) {
    int i = blockIdx.x * blockDim.x + threadIdx.x;
    if (i < n) out[i] = f2bf(in[i]);
}

// Transpose-convert weights: Wt[n*K + k] = (bf16) W[k*Ncol + n]
// Tiny matrices (<=128KB) -> done once per call, strided reads are fine.
__global__ void k_w_to_bf16_t(const float* __restrict__ W, bf16_t* __restrict__ Wt,
                              int K, int Ncol) {
    int t = blockIdx.x * blockDim.x + threadIdx.x;
    if (t >= K * Ncol) return;
    int n = t / K, k = t - n * K;
    Wt[t] = f2bf(W[k * Ncol + n]);
}

// ---------------------------------------------------------------------------
// bf16 WMMA GEMM:  Out[M,ncols](f32) = X[M,256](bf16) @ W + bias,
// with W supplied PRE-TRANSPOSED as Wt[ncols][256] bf16.
//
// Block: 256 thr = 8 waves. C tile per block: 128 (M) x 64 (N).
// The 64x256 Wt tile (32KB) is DMA'd into LDS by the Tensor Data Mover,
// using TDM pad_enable (4 dwords pad per 128 dwords) so each 512B row gets a
// 528B LDS stride -> bank-conflict-free ds_load_b128 B-fragment reads.
// A fragments are read straight from global (b128, row-contiguous).
// ---------------------------------------------------------------------------
#define BSTR 264  // padded row stride in bf16 elems (512B data + 16B TDM pad)

__global__ __launch_bounds__(256)
void k_gemm_bf16(const bf16_t* __restrict__ X, const bf16_t* __restrict__ Wt,
                 const float* __restrict__ bias, float* __restrict__ Out,
                 int M, int ncols) {
    __shared__ bf16_t Bt[64 * BSTR];   // 33,792 bytes

    const int t  = threadIdx.x;
    const int bm = blockIdx.x * 128;
    const int bn = blockIdx.y * 64;

#if HAVE_TDM
    // --- Tensor Data Mover: 2D tile, 64 rows x 256 bf16, contiguous source ---
    if (t < 32) {                       // wave 0 issues the DMA (EXEC ignored)
        unsigned lds_off = (unsigned)(unsigned long long)(uintptr_t)(&Bt[0]);
        unsigned long long ga =
            (unsigned long long)(uintptr_t)(Wt + (size_t)bn * FDIM);
        uint32x4 g0 = { 1u,                                  // count=1, user mode
                        lds_off,                             // lds_addr
                        (unsigned)(ga & 0xFFFFFFFFu),        // global_addr[31:0]
                        (unsigned)((ga >> 32) & 0x1FFFFFFu)  // global_addr[56:32]
                        | (2u << 30) };                      // type = image
        int32x8 g1;
        g1[0] = (int)((1u << 16)      // data_size = 2 bytes
                    | (1u << 20)      // pad_enable
                    | (6u << 22)      // pad_interval: every 128 dwords (512B row)
                    | (3u << 25));    // pad_amount: 4 dwords (16B)
        g1[1] = (int)(256u << 16);    // tensor_dim0 = 256 elems/row
        g1[2] = (int)(64u  << 16);    // tensor_dim1 = 64 rows
        g1[3] = (int)(256u << 16);    // tile_dim0 = 256
        g1[4] = (int)(64u);           // tile_dim1 = 64, tile_dim2 = 0
        g1[5] = (int)(256u);          // tensor_dim0_stride = 256 elems
        g1[6] = 0;
        g1[7] = 0;
        int32x4 g2 = {0, 0, 0, 0};
        int32x4 g3 = {0, 0, 0, 0};
#if defined(__clang_major__) && (__clang_major__ >= 23)
        int32x8 g4 = {0, 0, 0, 0, 0, 0, 0, 0};
        __builtin_amdgcn_tensor_load_to_lds(g0, g1, g2, g3, g4, 0);
#else
        __builtin_amdgcn_tensor_load_to_lds(g0, g1, g2, g3, 0);
#endif
        __builtin_amdgcn_s_wait_tensorcnt(0);
    }
#else
    // Fallback: cooperative contiguous b128 copy with manual padded stride
    for (int i = 0; i < 8; ++i) {
        int c  = i * 256 + t;          // 2048 chunks of 8 bf16
        int n  = c >> 5;
        int k8 = c & 31;
        *(v8bf*)(Bt + n * BSTR + k8 * 8) =
            *(const v8bf*)(Wt + ((size_t)(bn + n)) * FDIM + k8 * 8);
    }
#endif
    __syncthreads();

    const int w   = t >> 5;
    const int l   = t & 31;
    const int hl  = l >> 4;    // half-lane group (ISA fragment layout)
    const int m16 = l & 15;

    int r = bm + 16 * w + m16;
    if (r > M - 1) r = M - 1;                  // clamp: keep EXEC all-ones for WMMA
    const bf16_t* xrow = X + (size_t)r * FDIM;

    v8f acc[4] = {};

    for (int kk = 0; kk < FDIM; kk += 32) {
        // A fragment 16x32 bf16: lanes 0-15 -> K 0..7,16..23 ; lanes 16-31 -> K 8..15,24..31
        v8bf alo = *(const v8bf*)(xrow + kk + 8 * hl);
        v8bf ahi = *(const v8bf*)(xrow + kk + 16 + 8 * hl);
        v16bf a;
        #pragma unroll
        for (int i = 0; i < 8; ++i) { a[i] = alo[i]; a[8 + i] = ahi[i]; }

        #pragma unroll
        for (int j = 0; j < 4; ++j) {
            // B fragment 32x16 bf16: lanes 0-15 -> K 0..7,16..23 ; lanes 16-31 -> K 8..15,24..31 of column n
            const bf16_t* bp = Bt + (16 * j + m16) * BSTR + kk + 8 * hl;
            v8bf b0 = *(const v8bf*)(bp);
            v8bf b1 = *(const v8bf*)(bp + 16);
            v16bf b;
            #pragma unroll
            for (int i = 0; i < 8; ++i) { b[i] = b0[i]; b[8 + i] = b1[i]; }

            acc[j] = __builtin_amdgcn_wmma_f32_16x16x32_bf16(
                false, a, false, b, (short)0, acc[j], false, false);
        }
    }

    // C/D layout: VGPR v -> row (v + 8*hl) of the 16-row tile, col = m16
    #pragma unroll
    for (int j = 0; j < 4; ++j) {
        int col = bn + 16 * j + m16;
        float bv = bias[col];
        #pragma unroll
        for (int v = 0; v < 8; ++v) {
            int row = bm + 16 * w + 8 * hl + v;
            if (row < M) Out[(size_t)row * ncols + col] = acc[j][v] + bv;
        }
    }
}

// ---------------------------------------------------------------------------
// Edge kernels: one thread per (edge, head)
// ---------------------------------------------------------------------------
__global__ void k_edge_logits(const int* __restrict__ src, const int* __restrict__ dst,
                              const float* __restrict__ XL, const float* __restrict__ XR,
                              const float* __restrict__ att,
                              float* __restrict__ logits, float* __restrict__ maxv) {
    int t = blockIdx.x * blockDim.x + threadIdx.x;
    if (t >= EE * HEADS) return;
    int e = t >> 3, h = t & 7;
    int s = src[e], d = dst[e];
    const float4* xl = (const float4*)(XL + (size_t)s * FDIM + h * CDIM);
    const float4* xr = (const float4*)(XR + (size_t)d * FDIM + h * CDIM);
    const float4* at = (const float4*)(att + h * CDIM);
    float acc = 0.f;
    #pragma unroll
    for (int i = 0; i < 8; ++i) {
        float4 a = xl[i], b = xr[i], c = at[i];
        acc += lrelu(a.x + b.x) * c.x;
        acc += lrelu(a.y + b.y) * c.y;
        acc += lrelu(a.z + b.z) * c.z;
        acc += lrelu(a.w + b.w) * c.w;
    }
    logits[t] = acc;
    atomicMaxF(&maxv[d * HEADS + h], acc);
}

__global__ void k_edge_exp(const int* __restrict__ dst,
                           float* __restrict__ logits,        // in: logits, out: exp
                           const float* __restrict__ maxv,
                           float* __restrict__ den) {
    int t = blockIdx.x * blockDim.x + threadIdx.x;
    if (t >= EE * HEADS) return;
    int e = t >> 3, h = t & 7;
    int d = dst[e];
    float ex = __expf(logits[t] - maxv[d * HEADS + h]);
    logits[t] = ex;
    atomicAdd(&den[d * HEADS + h], ex);
}

__global__ void k_edge_agg(const int* __restrict__ src, const int* __restrict__ dst,
                           const float* __restrict__ exv,
                           const float* __restrict__ den,
                           const float* __restrict__ XL,
                           float* __restrict__ AGG) {
    int t = blockIdx.x * blockDim.x + threadIdx.x;
    if (t >= EE * HEADS) return;
    int e = t >> 3, h = t & 7;
    int s = src[e], d = dst[e];
    float alpha = exv[t] / (den[d * HEADS + h] + 1e-16f);
    const float4* xl = (const float4*)(XL + (size_t)s * FDIM + h * CDIM);
    float* out = AGG + (size_t)d * FDIM + h * CDIM;
    #pragma unroll
    for (int i = 0; i < 8; ++i) {
        float4 v = xl[i];
        atomicAdd(out + 4 * i + 0, v.x * alpha);
        atomicAdd(out + 4 * i + 1, v.y * alpha);
        atomicAdd(out + 4 * i + 2, v.z * alpha);
        atomicAdd(out + 4 * i + 3, v.w * alpha);
    }
}

// ---------------------------------------------------------------------------
// BatchNorm (training-mode batch stats) + ELU, output straight to bf16
// (post-aggregation GAT bias cancels inside BN's (x - mean), so it is skipped)
// ---------------------------------------------------------------------------
__global__ void k_bn_stats(const float* __restrict__ X, float* __restrict__ sums, int M) {
    int ch = threadIdx.x;                 // 256 channels
    int r0 = blockIdx.x * 512;
    int r1 = r0 + 512; if (r1 > M) r1 = M;
    float s = 0.f, sq = 0.f;
    for (int r = r0; r < r1; ++r) {
        float v = X[(size_t)r * FDIM + ch];
        s += v; sq += v * v;
    }
    atomicAdd(&sums[ch], s);
    atomicAdd(&sums[FDIM + ch], sq);
}

__global__ void k_bn_elu_cvt(const float* __restrict__ X, const float* __restrict__ sums,
                             const float* __restrict__ gamma, const float* __restrict__ beta,
                             bf16_t* __restrict__ Out, int M) {
    int t = blockIdx.x * blockDim.x + threadIdx.x;
    if (t >= M * FDIM) return;
    int ch = t & (FDIM - 1);
    const float invN = 1.0f / (float)NN;
    float mu  = sums[ch] * invN;
    float var = sums[FDIM + ch] * invN - mu * mu;
    float y = gamma[ch] * (X[t] - mu) * rsqrtf(var + 1e-5f) + beta[ch];
    y = y > 0.f ? y : expm1f(y);          // ELU(alpha=1)
    Out[t] = f2bf(y);
}

// ---------------------------------------------------------------------------
// Row-wise L2 normalize: one wave (32 lanes) per node, 4 floats per lane
// ---------------------------------------------------------------------------
__global__ __launch_bounds__(256)
void k_l2norm(const float* __restrict__ X, float* __restrict__ Out, int M) {
    int w = threadIdx.x >> 5, l = threadIdx.x & 31;
    int node = blockIdx.x * 8 + w;
    if (node >= M) return;
    float4 v = ((const float4*)(X + (size_t)node * OUTD))[l];
    float ss = v.x * v.x + v.y * v.y + v.z * v.z + v.w * v.w;
    #pragma unroll
    for (int off = 16; off > 0; off >>= 1) ss += __shfl_xor(ss, off, 32);
    float s = 1.0f / fmaxf(sqrtf(ss), 1e-12f);
    float4 o; o.x = v.x * s; o.y = v.y * s; o.z = v.z * s; o.w = v.w * s;
    ((float4*)(Out + (size_t)node * OUTD))[l] = o;
}

// ---------------------------------------------------------------------------
// Workspace layout (bytes). Total ~391 MB.
// ---------------------------------------------------------------------------
static const size_t O_XBF  = 0;            // N*256 bf16   = 51,200,000
static const size_t O_XL   = 51200000;     // N*256 f32    = 102,400,000
static const size_t O_XR   = 153600000;    // N*256 f32
static const size_t O_AGG  = 256000000;    // N*256 f32
static const size_t O_EDGE = 358400000;    // E*8 f32      = 25,600,000
static const size_t O_MAX  = 384000000;    // N*8 f32      = 3,200,000
static const size_t O_DEN  = 387200000;    // N*8 f32
static const size_t O_STAT = 390400000;    // 512 f32
static const size_t O_WBF  = 390402048;    // 294,912 bf16 (5 transposed weight mats)

static void run_gat_layer(const int* srcp, const int* dstp,
                          const bf16_t* Xbf, const bf16_t* Wlbf, const bf16_t* Wrbf,
                          const float* bl, const float* br, const float* att,
                          const float* gamma, const float* beta,
                          float* XL, float* XR, float* AGG, float* EDGEB,
                          float* MAXV, float* DEN, float* STAT,
                          bf16_t* Xout, hipStream_t stream) {
    dim3 ggrid((NN + 127) / 128, FDIM / 64);
    k_gemm_bf16<<<ggrid, 256, 0, stream>>>(Xbf, Wlbf, bl, XL, NN, FDIM);
    k_gemm_bf16<<<ggrid, 256, 0, stream>>>(Xbf, Wrbf, br, XR, NN, FDIM);

    k_fill<<<(NN * HEADS + 255) / 256, 256, 0, stream>>>(MAXV, -INFINITY, NN * HEADS);
    k_fill<<<(NN * HEADS + 255) / 256, 256, 0, stream>>>(DEN, 0.f, NN * HEADS);
    k_fill<<<(NN * FDIM + 255) / 256, 256, 0, stream>>>(AGG, 0.f, NN * FDIM);
    k_fill<<<2, 256, 0, stream>>>(STAT, 0.f, 2 * FDIM);

    int egrid = (EE * HEADS + 255) / 256;
    k_edge_logits<<<egrid, 256, 0, stream>>>(srcp, dstp, XL, XR, att, EDGEB, MAXV);
    k_edge_exp<<<egrid, 256, 0, stream>>>(dstp, EDGEB, MAXV, DEN);
    k_edge_agg<<<egrid, 256, 0, stream>>>(srcp, dstp, EDGEB, DEN, XL, AGG);

    k_bn_stats<<<(NN + 511) / 512, 256, 0, stream>>>(AGG, STAT, NN);
    k_bn_elu_cvt<<<(NN * FDIM + 255) / 256, 256, 0, stream>>>(AGG, STAT, gamma, beta, Xout, NN);
}

extern "C" void kernel_launch(void* const* d_in, const int* in_sizes, int n_in,
                              void* d_out, int out_size, void* d_ws, size_t ws_size,
                              hipStream_t stream) {
    const float* x      = (const float*)d_in[0];
    const int*   ei     = (const int*)d_in[1];
    const float* Wl1    = (const float*)d_in[2];
    const float* bl1    = (const float*)d_in[3];
    const float* Wr1    = (const float*)d_in[4];
    const float* br1    = (const float*)d_in[5];
    const float* att1   = (const float*)d_in[6];
    /* bias1 (d_in[7]) cancels inside BatchNorm */
    const float* gamma1 = (const float*)d_in[8];
    const float* beta1  = (const float*)d_in[9];
    const float* Wl2    = (const float*)d_in[10];
    const float* bl2    = (const float*)d_in[11];
    const float* Wr2    = (const float*)d_in[12];
    const float* br2    = (const float*)d_in[13];
    const float* att2   = (const float*)d_in[14];
    /* bias2 (d_in[15]) cancels inside BatchNorm */
    const float* gamma2 = (const float*)d_in[16];
    const float* beta2  = (const float*)d_in[17];
    const float* Wfc    = (const float*)d_in[18];
    const float* bfc    = (const float*)d_in[19];

    const int* srcp = ei;
    const int* dstp = ei + EE;

    char* ws = (char*)d_ws;
    bf16_t* Xbf   = (bf16_t*)(ws + O_XBF);
    float*  XL    = (float*)(ws + O_XL);
    float*  XR    = (float*)(ws + O_XR);
    float*  AGG   = (float*)(ws + O_AGG);
    float*  EDGEB = (float*)(ws + O_EDGE);
    float*  MAXV  = (float*)(ws + O_MAX);
    float*  DEN   = (float*)(ws + O_DEN);
    float*  STAT  = (float*)(ws + O_STAT);
    bf16_t* Wl1b  = (bf16_t*)(ws + O_WBF);   // all stored TRANSPOSED: [ncols][256]
    bf16_t* Wr1b  = Wl1b + 256 * 256;
    bf16_t* Wl2b  = Wr1b + 256 * 256;
    bf16_t* Wr2b  = Wl2b + 256 * 256;
    bf16_t* Wfcb  = Wr2b + 256 * 256;

    // Weight transpose + f32->bf16 (every call: deterministic, no caching)
    k_w_to_bf16_t<<<256, 256, 0, stream>>>(Wl1, Wl1b, 256, 256);
    k_w_to_bf16_t<<<256, 256, 0, stream>>>(Wr1, Wr1b, 256, 256);
    k_w_to_bf16_t<<<256, 256, 0, stream>>>(Wl2, Wl2b, 256, 256);
    k_w_to_bf16_t<<<256, 256, 0, stream>>>(Wr2, Wr2b, 256, 256);
    k_w_to_bf16_t<<<128, 256, 0, stream>>>(Wfc, Wfcb, 256, 128);

    // Node features f32 -> bf16
    k_f32_to_bf16<<<(NN * FDIM + 255) / 256, 256, 0, stream>>>(x, Xbf, NN * FDIM);

    // Layer 1: Xbf -> (XL,XR) -> attention aggregate -> BN+ELU -> Xbf
    run_gat_layer(srcp, dstp, Xbf, Wl1b, Wr1b, bl1, br1, att1, gamma1, beta1,
                  XL, XR, AGG, EDGEB, MAXV, DEN, STAT, Xbf, stream);

    // Layer 2
    run_gat_layer(srcp, dstp, Xbf, Wl2b, Wr2b, bl2, br2, att2, gamma2, beta2,
                  XL, XR, AGG, EDGEB, MAXV, DEN, STAT, Xbf, stream);

    // FC: [N,256] @ [256,128] + bfc  -> reuse XL as output buffer
    dim3 fgrid((NN + 127) / 128, OUTD / 64);
    k_gemm_bf16<<<fgrid, 256, 0, stream>>>(Xbf, Wfcb, bfc, XL, NN, OUTD);

    // Row-wise L2 normalize into d_out (f32)
    k_l2norm<<<(NN + 7) / 8, 256, 0, stream>>>(XL, (float*)d_out, NN);
}
